// ContextTransformer_18751827214945
// MI455X (gfx1250) — matrix operand order repaired and checked by reading, not compiled, not run on hardware
//
#include <hip/hip_runtime.h>
#include <hip/hip_bf16.h>

// ---------------------------------------------------------------------------
// CDNA5 (gfx1250) ContextTransformer forward.
// Fast GEMM: v_wmma_f32_16x16x32_f16, 128x64 tile, double-buffered LDS,
// global_load_b128 staging, ds_load_b128 fragments, 1 barrier / K-step.
// Generic GEMM handles ragged shapes (K=161/2/1, N=129/32/16, M=511).
// Attention exploits the mask: only j in [0,CTX) and j==T-1 are unmasked.
// ---------------------------------------------------------------------------

typedef __attribute__((ext_vector_type(16))) _Float16 v16h;
typedef __attribute__((ext_vector_type(8)))  _Float16 v8h;
typedef __attribute__((ext_vector_type(8)))  float    v8f;
typedef __attribute__((ext_vector_type(4)))  float    v4f;

#define BATCH  32
#define SEQ    256
#define CTXN   10
#define DMODEL 512
#define NHEAD  8
#define DHEAD  64
#define DPFF   2048
#define DIN    161
#define NEXP   16

#define BM 128
#define BN 64
#define BK 32
#define LDA 40      // padded LDS row stride in halves (80B, 16B aligned)
#define LDB 40

// ---- shared epilogue --------------------------------------------------------
__device__ __forceinline__ float epilogue_val(
    float val, int gr, int gc,
    const float* __restrict__ bias, int act, float alpha,
    const float* __restrict__ rowScale, int scaleStride, int scaleOff)
{
    if (bias) val += bias[gc];
    if (act == 1)      val = val >= 0.0f ? val : alpha * val;
    else if (act == 2) val = fmaxf(val, 0.0f);
    if (rowScale) val *= rowScale[(size_t)gr * scaleStride + scaleOff];
    return val;
}

// ---------------------------------------------------------------------------
// FAST GEMM: requires M%128==0, N%64==0, K%32==0.  Double-buffered LDS.
// ---------------------------------------------------------------------------
__global__ __launch_bounds__(256) void gemm_wmma_fast_kernel(
    const float* __restrict__ A, const float* __restrict__ Bw,
    const float* __restrict__ bias,
    const float* __restrict__ rowScale, int scaleStride, int scaleOff,
    float* __restrict__ C, int M, int N, int K,
    int accumulate, int act, const float* __restrict__ alphaPtr)
{
    __shared__ _Float16 As[2][BM * LDA];
    __shared__ _Float16 BsT[2][BN * LDB];

    const int tid  = threadIdx.x;
    const int wave = tid >> 5;
    const int lane = tid & 31;
    const int blockRow = blockIdx.y * BM;
    const int blockCol = blockIdx.x * BN;

    const int warpM = wave >> 1;
    const int warpN = wave & 1;
    const int rBase = warpM * 32;
    const int cBase = warpN * 32;
    const int laneHalf = lane >> 4;
    const int lane16   = lane & 15;

    // staging coordinates
    const int ar = tid >> 1;               // A row 0..127
    const int ak = (tid & 1) * 16;         // A k  0 or 16
    const int brk = tid >> 3;              // B k row 0..31
    const int bc  = (tid & 7) * 8;         // B col 0..56
    const float* aPtr = A  + (size_t)(blockRow + ar) * K + ak;
    const float* bPtr = Bw + (size_t)brk * N + blockCol + bc;

    v8f acc00 = {}, acc01 = {}, acc10 = {}, acc11 = {};

    const int ktiles = K >> 5;

    // ---- stage tile 0 into buffer 0 ----
    {
        v4f a0 = *(const v4f*)(aPtr);
        v4f a1 = *(const v4f*)(aPtr + 4);
        v4f a2 = *(const v4f*)(aPtr + 8);
        v4f a3 = *(const v4f*)(aPtr + 12);
        v4f b0 = *(const v4f*)(bPtr);
        v4f b1 = *(const v4f*)(bPtr + 4);
        v8h h0, h1;
        #pragma unroll
        for (int j = 0; j < 4; ++j) {
            h0[j] = (_Float16)a0[j]; h0[j + 4] = (_Float16)a1[j];
            h1[j] = (_Float16)a2[j]; h1[j + 4] = (_Float16)a3[j];
        }
        *(v8h*)&As[0][ar * LDA + ak]     = h0;
        *(v8h*)&As[0][ar * LDA + ak + 8] = h1;
        #pragma unroll
        for (int j = 0; j < 4; ++j) {
            BsT[0][(bc + j)     * LDB + brk] = (_Float16)b0[j];
            BsT[0][(bc + j + 4) * LDB + brk] = (_Float16)b1[j];
        }
    }
    __syncthreads();

    for (int kt = 0; kt < ktiles; ++kt) {
        const int p = kt & 1;
        const bool more = (kt + 1 < ktiles);

        // ---- issue next tile's global loads early (latency overlap) ----
        v4f na0, na1, na2, na3, nb0, nb1;
        if (more) {
            const float* ap = aPtr + (kt + 1) * BK;
            const float* bp = bPtr + (size_t)(kt + 1) * BK * N;
            na0 = *(const v4f*)(ap);
            na1 = *(const v4f*)(ap + 4);
            na2 = *(const v4f*)(ap + 8);
            na3 = *(const v4f*)(ap + 12);
            nb0 = *(const v4f*)(bp);
            nb1 = *(const v4f*)(bp + 4);
        }

        // ---- fragments from buffer p: 2x ds_load_b128 each ----
        const v8h* ar0 = (const v8h*)&As[p][(rBase + lane16) * LDA];
        const v8h* ar1 = (const v8h*)&As[p][(rBase + 16 + lane16) * LDA];
        v16h af0 = __builtin_shufflevector(ar0[laneHalf], ar0[laneHalf + 2],
                     0,1,2,3,4,5,6,7,8,9,10,11,12,13,14,15);
        v16h af1 = __builtin_shufflevector(ar1[laneHalf], ar1[laneHalf + 2],
                     0,1,2,3,4,5,6,7,8,9,10,11,12,13,14,15);
        const v8h* bc0 = (const v8h*)&BsT[p][(cBase + lane16) * LDB];
        const v8h* bc1 = (const v8h*)&BsT[p][(cBase + 16 + lane16) * LDB];
        v16h bf0 = __builtin_shufflevector(bc0[laneHalf], bc0[laneHalf + 2],
                     0,1,2,3,4,5,6,7,8,9,10,11,12,13,14,15);
        v16h bf1 = __builtin_shufflevector(bc1[laneHalf], bc1[laneHalf + 2],
                     0,1,2,3,4,5,6,7,8,9,10,11,12,13,14,15);

        acc00 = __builtin_amdgcn_wmma_f32_16x16x32_f16(false, af0, false, bf0,
                                                       (short)0, acc00, false, false);
        acc01 = __builtin_amdgcn_wmma_f32_16x16x32_f16(false, af0, false, bf1,
                                                       (short)0, acc01, false, false);
        acc10 = __builtin_amdgcn_wmma_f32_16x16x32_f16(false, af1, false, bf0,
                                                       (short)0, acc10, false, false);
        acc11 = __builtin_amdgcn_wmma_f32_16x16x32_f16(false, af1, false, bf1,
                                                       (short)0, acc11, false, false);

        // ---- store next tile into the other buffer ----
        if (more) {
            const int q = 1 - p;
            v8h h0, h1;
            #pragma unroll
            for (int j = 0; j < 4; ++j) {
                h0[j] = (_Float16)na0[j]; h0[j + 4] = (_Float16)na1[j];
                h1[j] = (_Float16)na2[j]; h1[j + 4] = (_Float16)na3[j];
            }
            *(v8h*)&As[q][ar * LDA + ak]     = h0;
            *(v8h*)&As[q][ar * LDA + ak + 8] = h1;
            #pragma unroll
            for (int j = 0; j < 4; ++j) {
                BsT[q][(bc + j)     * LDB + brk] = (_Float16)nb0[j];
                BsT[q][(bc + j + 4) * LDB + brk] = (_Float16)nb1[j];
            }
        }
        __syncthreads();
    }

    // ---- epilogue: full tiles, no bounds checks ----
    float alpha = (act == 1 && alphaPtr) ? alphaPtr[0] : 0.0f;
    #pragma unroll
    for (int mt = 0; mt < 2; ++mt) {
        #pragma unroll
        for (int nt = 0; nt < 2; ++nt) {
            const v8f& acc = mt == 0 ? (nt == 0 ? acc00 : acc01)
                                     : (nt == 0 ? acc10 : acc11);
            #pragma unroll
            for (int v = 0; v < 8; ++v) {
                int gr = blockRow + rBase + mt * 16 + v + laneHalf * 8;
                int gc = blockCol + cBase + nt * 16 + lane16;
                float val = epilogue_val(acc[v], gr, gc, bias, act, alpha,
                                         rowScale, scaleStride, scaleOff);
                size_t o = (size_t)gr * N + gc;
                C[o] = accumulate ? (C[o] + val) : val;
            }
        }
    }
}

// ---------------------------------------------------------------------------
// GENERIC GEMM: any M,N,K (clamped branch-free staging on edges).
// ---------------------------------------------------------------------------
__global__ __launch_bounds__(256) void gemm_wmma_kernel(
    const float* __restrict__ A, const float* __restrict__ Bw,
    const float* __restrict__ bias,
    const float* __restrict__ rowScale, int scaleStride, int scaleOff,
    float* __restrict__ C, int M, int N, int K,
    int accumulate, int act, const float* __restrict__ alphaPtr)
{
    __shared__ _Float16 As[BM * LDA];
    __shared__ _Float16 BsT[BN * LDB];

    const int tid  = threadIdx.x;
    const int wave = tid >> 5;
    const int lane = tid & 31;
    const int blockRow = blockIdx.y * BM;
    const int blockCol = blockIdx.x * BN;

    const int warpM = wave >> 1;
    const int warpN = wave & 1;
    const int rBase = warpM * 32;
    const int cBase = warpN * 32;
    const int laneHalf = lane >> 4;
    const int lane16   = lane & 15;

    v8f acc00 = {}, acc01 = {}, acc10 = {}, acc11 = {};

    for (int k0 = 0; k0 < K; k0 += BK) {
        // clamped, branch-free staging (edge-tolerant)
        #pragma unroll
        for (int s = 0; s < 16; ++s) {
            int idx = tid * 16 + s;
            int r  = idx >> 5;
            int kk = idx & 31;
            int gr = blockRow + r;
            int gk = k0 + kk;
            int cr = gr < M ? gr : M - 1;
            int ck = gk < K ? gk : K - 1;
            float av = A[(size_t)cr * K + ck];
            As[r * LDA + kk] = (_Float16)((gr < M && gk < K) ? av : 0.0f);
        }
        #pragma unroll
        for (int s = 0; s < 8; ++s) {
            int idx = tid * 8 + s;
            int kk = idx >> 6;
            int c  = idx & 63;
            int gk = k0 + kk;
            int gc = blockCol + c;
            int ckk = gk < K ? gk : K - 1;
            int cc  = gc < N ? gc : N - 1;
            float bv = Bw[(size_t)ckk * N + cc];
            BsT[c * LDB + kk] = (_Float16)((gk < K && gc < N) ? bv : 0.0f);
        }
        __syncthreads();

        const v8h* ar0 = (const v8h*)&As[(rBase + lane16) * LDA];
        const v8h* ar1 = (const v8h*)&As[(rBase + 16 + lane16) * LDA];
        v16h af0 = __builtin_shufflevector(ar0[laneHalf], ar0[laneHalf + 2],
                     0,1,2,3,4,5,6,7,8,9,10,11,12,13,14,15);
        v16h af1 = __builtin_shufflevector(ar1[laneHalf], ar1[laneHalf + 2],
                     0,1,2,3,4,5,6,7,8,9,10,11,12,13,14,15);
        const v8h* bc0 = (const v8h*)&BsT[(cBase + lane16) * LDB];
        const v8h* bc1 = (const v8h*)&BsT[(cBase + 16 + lane16) * LDB];
        v16h bf0 = __builtin_shufflevector(bc0[laneHalf], bc0[laneHalf + 2],
                     0,1,2,3,4,5,6,7,8,9,10,11,12,13,14,15);
        v16h bf1 = __builtin_shufflevector(bc1[laneHalf], bc1[laneHalf + 2],
                     0,1,2,3,4,5,6,7,8,9,10,11,12,13,14,15);

        acc00 = __builtin_amdgcn_wmma_f32_16x16x32_f16(false, af0, false, bf0,
                                                       (short)0, acc00, false, false);
        acc01 = __builtin_amdgcn_wmma_f32_16x16x32_f16(false, af0, false, bf1,
                                                       (short)0, acc01, false, false);
        acc10 = __builtin_amdgcn_wmma_f32_16x16x32_f16(false, af1, false, bf0,
                                                       (short)0, acc10, false, false);
        acc11 = __builtin_amdgcn_wmma_f32_16x16x32_f16(false, af1, false, bf1,
                                                       (short)0, acc11, false, false);
        __syncthreads();
    }

    float alpha = (act == 1 && alphaPtr) ? alphaPtr[0] : 0.0f;
    #pragma unroll
    for (int mt = 0; mt < 2; ++mt) {
        #pragma unroll
        for (int nt = 0; nt < 2; ++nt) {
            const v8f& acc = mt == 0 ? (nt == 0 ? acc00 : acc01)
                                     : (nt == 0 ? acc10 : acc11);
            #pragma unroll
            for (int v = 0; v < 8; ++v) {
                int gr = blockRow + rBase + mt * 16 + v + laneHalf * 8;
                int gc = blockCol + cBase + nt * 16 + lane16;
                if (gr < M && gc < N) {
                    float val = epilogue_val(acc[v], gr, gc, bias, act, alpha,
                                             rowScale, scaleStride, scaleOff);
                    size_t o = (size_t)gr * N + gc;
                    C[o] = accumulate ? (C[o] + val) : val;
                }
            }
        }
    }
}

// ---------------------------------------------------------------------------
// Elementwise / small kernels
// ---------------------------------------------------------------------------
__global__ void build_xin_kernel(const float* __restrict__ motion,
                                 const float* __restrict__ phase,
                                 float* __restrict__ xin)
{
    int idx = blockIdx.x * blockDim.x + threadIdx.x;
    int total = BATCH * SEQ * DIN;
    if (idx >= total) return;
    int row = idx / DIN;
    int c   = idx - row * DIN;
    int t   = row & (SEQ - 1);
    float dm = (t < CTXN || t == SEQ - 1) ? 1.0f : 0.0f;
    float v;
    if (c < 128)       v = motion[(size_t)row * 128 + c] * dm;
    else if (c == 128) v = dm;
    else               v = phase[(size_t)row * 32 + (c - 129)] * dm;
    xin[idx] = v;
}

__global__ void kf_build_kernel(float* __restrict__ kf)
{
    int t = blockIdx.x * blockDim.x + threadIdx.x;
    if (t >= SEQ) return;
    kf[t * 2 + 0] = (float)t - (float)(CTXN - 1);
    kf[t * 2 + 1] = (float)(SEQ - 1 - t);
}

__global__ void relidx_build_kernel(float* __restrict__ ri)
{
    int l = blockIdx.x * blockDim.x + threadIdx.x;
    if (l >= 2 * SEQ - 1) return;
    ri[l] = (float)(l - (SEQ - 1));
}

__global__ void add_kp_kernel(float* __restrict__ x, const float* __restrict__ kp)
{
    int idx = blockIdx.x * blockDim.x + threadIdx.x;
    int total = BATCH * SEQ * DMODEL;
    if (idx >= total) return;
    x[idx] += kp[idx % (SEQ * DMODEL)];
}

__global__ __launch_bounds__(256) void ln_kernel(const float* __restrict__ x,
                                                 const float* __restrict__ g,
                                                 const float* __restrict__ b,
                                                 float* __restrict__ out)
{
    __shared__ float red[256];
    int row = blockIdx.x;
    const float* xr = x + (size_t)row * DMODEL;
    int c = threadIdx.x;
    float v0 = xr[c], v1 = xr[c + 256];
    red[c] = v0 + v1;
    __syncthreads();
    for (int o = 128; o > 0; o >>= 1) {
        if (c < o) red[c] += red[c + o];
        __syncthreads();
    }
    float mu = red[0] / (float)DMODEL;
    __syncthreads();
    float d0 = v0 - mu, d1 = v1 - mu;
    red[c] = d0 * d0 + d1 * d1;
    __syncthreads();
    for (int o = 128; o > 0; o >>= 1) {
        if (c < o) red[c] += red[c + o];
        __syncthreads();
    }
    float r = rsqrtf(red[0] / (float)DMODEL + 1e-5f);
    float* orow = out + (size_t)row * DMODEL;
    orow[c]       = d0 * r * g[c]       + b[c];
    orow[c + 256] = d1 * r * g[c + 256] + b[c + 256];
}

// Fused masked attention: only j in [0,CTX) and j==T-1 survive the mask.
__global__ __launch_bounds__(256) void attention_kernel(
    const float* __restrict__ Q, const float* __restrict__ K,
    const float* __restrict__ V, const float* __restrict__ REL,
    float* __restrict__ AO)
{
    int bi = blockIdx.x;
    int b = bi >> 8;
    int i = bi & (SEQ - 1);
    int h = threadIdx.x >> 5;
    int lane = threadIdx.x & 31;
    int d0 = lane * 2, d1 = d0 + 1;

    size_t qo = ((size_t)(b * SEQ + i)) * DMODEL + h * DHEAD;
    float q0 = Q[qo + d0], q1 = Q[qo + d1];

    float s[CTXN + 1];
    #pragma unroll
    for (int jj = 0; jj <= CTXN; ++jj) {
        int j = (jj < CTXN) ? jj : (SEQ - 1);
        size_t ko = ((size_t)(b * SEQ + j)) * DMODEL + h * DHEAD;
        int l = j - i + (SEQ - 1);
        float k0 = K[ko + d0] + REL[(size_t)l * DHEAD + d0];
        float k1 = K[ko + d1] + REL[(size_t)l * DHEAD + d1];
        float part = q0 * k0 + q1 * k1;
        for (int o = 16; o > 0; o >>= 1) part += __shfl_xor(part, o);
        s[jj] = part * 0.125f;   // 1/sqrt(64)
    }
    float m = s[0];
    #pragma unroll
    for (int jj = 1; jj <= CTXN; ++jj) m = fmaxf(m, s[jj]);
    float e[CTXN + 1];
    float sum = 0.0f;
    #pragma unroll
    for (int jj = 0; jj <= CTXN; ++jj) { e[jj] = __expf(s[jj] - m); sum += e[jj]; }
    float inv = 1.0f / sum;
    float o0 = 0.0f, o1 = 0.0f;
    #pragma unroll
    for (int jj = 0; jj <= CTXN; ++jj) {
        int j = (jj < CTXN) ? jj : (SEQ - 1);
        size_t vo = ((size_t)(b * SEQ + j)) * DMODEL + h * DHEAD;
        float w = e[jj] * inv;
        o0 += w * V[vo + d0];
        o1 += w * V[vo + d1];
    }
    AO[qo + d0] = o0;
    AO[qo + d1] = o1;
}

__global__ void gate_softmax_kernel(float* __restrict__ ew, int rows)
{
    int r = blockIdx.x * blockDim.x + threadIdx.x;
    if (r >= rows) return;
    float* p = ew + (size_t)r * NEXP;
    float m = p[0];
    #pragma unroll
    for (int i = 1; i < NEXP; ++i) m = fmaxf(m, p[i]);
    float e[NEXP], s = 0.0f;
    #pragma unroll
    for (int i = 0; i < NEXP; ++i) { e[i] = __expf(p[i] - m); s += e[i]; }
    float inv = 1.0f / s;
    #pragma unroll
    for (int i = 0; i < NEXP; ++i) p[i] = e[i] * inv;
}

__global__ void prelu_ip_kernel(float* __restrict__ x, int n,
                                const float* __restrict__ alphaPtr)
{
    int idx = blockIdx.x * blockDim.x + threadIdx.x;
    if (idx >= n) return;
    float a = alphaPtr[0];
    float v = x[idx];
    x[idx] = v >= 0.0f ? v : a * v;
}

__global__ void finalize_kernel(const float* __restrict__ m2,
                                const float* __restrict__ ph,
                                float* __restrict__ out)
{
    const int BT = BATCH * SEQ;
    int idx = blockIdx.x * blockDim.x + threadIdx.x;
    int total = BT * 161;
    if (idx >= total) return;
    if (idx < BT * 128) {
        int r = idx >> 7, c = idx & 127;
        out[idx] = m2[(size_t)r * 129 + c];
    } else if (idx < BT * 160) {
        int k = idx - BT * 128;
        out[BT * 128 + k] = ph[k];
    } else {
        int r = idx - BT * 160;
        float v = m2[(size_t)r * 129 + 128];
        out[BT * 160 + r] = 1.0f / (1.0f + __expf(-v));
    }
}

// ---------------------------------------------------------------------------
// Host orchestration
// ---------------------------------------------------------------------------
extern "C" void kernel_launch(void* const* d_in, const int* in_sizes, int n_in,
                              void* d_out, int out_size, void* d_ws, size_t ws_size,
                              hipStream_t stream)
{
    (void)in_sizes; (void)n_in; (void)out_size; (void)ws_size;

    const float* motion   = (const float*)d_in[0];
    const float* phase    = (const float*)d_in[1];
    const float* enc_w1   = (const float*)d_in[2];
    const float* enc_b1   = (const float*)d_in[3];
    const float* enc_a1   = (const float*)d_in[4];
    const float* enc_w2   = (const float*)d_in[5];
    const float* enc_b2   = (const float*)d_in[6];
    const float* enc_a2   = (const float*)d_in[7];
    const float* kf_w1    = (const float*)d_in[8];
    const float* kf_b1    = (const float*)d_in[9];
    const float* kf_a1    = (const float*)d_in[10];
    const float* kf_w2    = (const float*)d_in[11];
    const float* kf_b2    = (const float*)d_in[12];
    const float* rp_w1    = (const float*)d_in[13];
    const float* rp_b1    = (const float*)d_in[14];
    const float* rp_a1    = (const float*)d_in[15];
    const float* rp_w2    = (const float*)d_in[16];
    const float* rp_b2    = (const float*)d_in[17];
    const float* att_ln_g = (const float*)d_in[18];
    const float* att_ln_b = (const float*)d_in[19];
    const float* wq       = (const float*)d_in[20];
    const float* wk       = (const float*)d_in[21];
    const float* wv       = (const float*)d_in[22];
    const float* wo       = (const float*)d_in[23];
    const float* pff_ln_g = (const float*)d_in[24];
    const float* pff_ln_b = (const float*)d_in[25];
    const float* pff_w1   = (const float*)d_in[26];
    const float* pff_b1   = (const float*)d_in[27];
    const float* pff_w2   = (const float*)d_in[28];
    const float* pff_b2   = (const float*)d_in[29];
    const float* fin_g    = (const float*)d_in[30];
    const float* fin_b    = (const float*)d_in[31];
    const float* pd_w1    = (const float*)d_in[32];
    const float* pd_b1    = (const float*)d_in[33];
    const float* pd_a1    = (const float*)d_in[34];
    const float* pd_w2    = (const float*)d_in[35];
    const float* pd_b2    = (const float*)d_in[36];
    const float* g_w1     = (const float*)d_in[37];
    const float* g_b1     = (const float*)d_in[38];
    const float* g_a1     = (const float*)d_in[39];
    const float* g_w2     = (const float*)d_in[40];
    const float* g_b2     = (const float*)d_in[41];
    const float* g_a2     = (const float*)d_in[42];
    const float* g_w3     = (const float*)d_in[43];
    const float* g_b3     = (const float*)d_in[44];
    const float* md_w0    = (const float*)d_in[45];
    const float* md_b0    = (const float*)d_in[46];
    const float* md_a0    = (const float*)d_in[47];
    const float* md_w1    = (const float*)d_in[48];
    const float* md_b1    = (const float*)d_in[49];
    const float* md_a1    = (const float*)d_in[50];
    const float* md_w2    = (const float*)d_in[51];
    const float* md_b2    = (const float*)d_in[52];

    float* out = (float*)d_out;

    // ---- workspace carving (floats, 16B-aligned blocks) -------------------
    const size_t BT = (size_t)BATCH * SEQ;   // 8192
    float* ws = (float*)d_ws;
    size_t off = 0;
    auto alloc = [&](size_t n) {
        float* p = ws + off;
        off += (n + 3) & ~(size_t)3;
        return p;
    };

    float* XIN = alloc(BT * DIN);
    float* X   = alloc(BT * DMODEL);
    float* H   = alloc(BT * DMODEL);
    float* S   = alloc(BT * DPFF);           // 64 MB, time-shared
    float* Q   = S;
    float* Kb  = S + BT * DMODEL;
    float* V   = S + 2 * BT * DMODEL;
    float* AO  = S + 3 * BT * DMODEL;
    float* PFF = S;
    float* PD  = S;
    float* M0  = S + BT * DMODEL;
    float* M1  = S + 2 * BT * DMODEL;
    float* G1  = S + 3 * BT * DMODEL;
    float* G2  = G1 + BT * 128;
    float* KF   = alloc(SEQ * 2);
    float* KPA  = alloc(SEQ * DMODEL);
    float* KP   = alloc(SEQ * DMODEL);
    float* RELI = alloc(2 * SEQ - 1);
    float* RELA = alloc((2 * SEQ - 1) * DHEAD);
    float* REL  = alloc((2 * SEQ - 1) * DHEAD);
    float* PH   = alloc(BT * 32);
    float* EW   = alloc(BT * NEXP);
    float* M2   = alloc(BT * 129);

    auto gemm = [&](const float* A, const float* Bw, const float* bias, float* C,
                    int M, int N, int K, int accumulate, int act,
                    const float* alpha,
                    const float* rowScale = nullptr, int sStride = 0, int sOff = 0) {
        dim3 grid((N + BN - 1) / BN, (M + BM - 1) / BM);
        bool fast = (M % BM == 0) && (N % BN == 0) && (K % BK == 0) && (K >= BK);
        if (fast)
            gemm_wmma_fast_kernel<<<grid, dim3(256), 0, stream>>>(
                A, Bw, bias, rowScale, sStride, sOff,
                C, M, N, K, accumulate, act, alpha);
        else
            gemm_wmma_kernel<<<grid, dim3(256), 0, stream>>>(
                A, Bw, bias, rowScale, sStride, sOff,
                C, M, N, K, accumulate, act, alpha);
    };
    auto ew_blocks = [](size_t n) { return (unsigned)((n + 255) / 256); };

    // ---- preprocessing ----------------------------------------------------
    build_xin_kernel<<<ew_blocks(BT * DIN), 256, 0, stream>>>(motion, phase, XIN);
    kf_build_kernel<<<1, SEQ, 0, stream>>>(KF);
    relidx_build_kernel<<<2, 256, 0, stream>>>(RELI);

    gemm(KF,  kf_w1, kf_b1, KPA, SEQ, DMODEL, 2,       0, 1, kf_a1);
    gemm(KPA, kf_w2, kf_b2, KP,  SEQ, DMODEL, DMODEL,  0, 0, nullptr);

    gemm(RELI, rp_w1, rp_b1, RELA, 2 * SEQ - 1, DHEAD, 1,     0, 1, rp_a1);
    gemm(RELA, rp_w2, rp_b2, REL,  2 * SEQ - 1, DHEAD, DHEAD, 0, 0, nullptr);

    gemm(XIN, enc_w1, enc_b1, H, (int)BT, DMODEL, DIN,    0, 1, enc_a1);
    gemm(H,   enc_w2, enc_b2, X, (int)BT, DMODEL, DMODEL, 0, 1, enc_a2);
    add_kp_kernel<<<ew_blocks(BT * DMODEL), 256, 0, stream>>>(X, KP);

    // ---- transformer layers ----------------------------------------------
    for (int i = 0; i < 6; ++i) {
        const size_t wOff = (size_t)i * DMODEL * DMODEL;
        const size_t lOff = (size_t)i * DMODEL;

        ln_kernel<<<(unsigned)BT, 256, 0, stream>>>(X, att_ln_g + lOff, att_ln_b + lOff, H);
        gemm(H, wq + wOff, nullptr, Q,  (int)BT, DMODEL, DMODEL, 0, 0, nullptr);
        gemm(H, wk + wOff, nullptr, Kb, (int)BT, DMODEL, DMODEL, 0, 0, nullptr);
        gemm(H, wv + wOff, nullptr, V,  (int)BT, DMODEL, DMODEL, 0, 0, nullptr);
        attention_kernel<<<(unsigned)BT, 256, 0, stream>>>(Q, Kb, V, REL, AO);
        gemm(AO, wo + wOff, nullptr, X, (int)BT, DMODEL, DMODEL, 1, 0, nullptr);

        ln_kernel<<<(unsigned)BT, 256, 0, stream>>>(X, pff_ln_g + lOff, pff_ln_b + lOff, H);
        gemm(H, pff_w1 + (size_t)i * DMODEL * DPFF, pff_b1 + (size_t)i * DPFF,
             PFF, (int)BT, DPFF, DMODEL, 0, 2, nullptr);
        gemm(PFF, pff_w2 + (size_t)i * DPFF * DMODEL, pff_b2 + lOff,
             X, (int)BT, DMODEL, DPFF, 1, 0, nullptr);
    }

    // ---- decoder ----------------------------------------------------------
    ln_kernel<<<(unsigned)BT, 256, 0, stream>>>(X, fin_g, fin_b, H);

    gemm(H,  pd_w1, pd_b1, PD, (int)BT, DMODEL, DMODEL, 0, 1, pd_a1);
    gemm(PD, pd_w2, pd_b2, PH, (int)BT, 32,     DMODEL, 0, 0, nullptr);

    gemm(PH, g_w1, g_b1, G1, (int)BT, 128, 32,  0, 1, g_a1);
    gemm(G1, g_w2, g_b2, G2, (int)BT, 128, 128, 0, 1, g_a2);
    gemm(G2, g_w3, g_b3, EW, (int)BT, NEXP, 128, 0, 0, nullptr);
    gate_softmax_kernel<<<ew_blocks(BT), 256, 0, stream>>>(EW, (int)BT);

    for (int e = 0; e < NEXP; ++e)
        gemm(H, md_w0 + (size_t)e * DMODEL * DMODEL, md_b0 + (size_t)e * DMODEL,
             M0, (int)BT, DMODEL, DMODEL, e > 0, 0, nullptr, EW, NEXP, e);
    prelu_ip_kernel<<<ew_blocks(BT * DMODEL), 256, 0, stream>>>(M0, (int)(BT * DMODEL), md_a0);

    for (int e = 0; e < NEXP; ++e)
        gemm(M0, md_w1 + (size_t)e * DMODEL * DMODEL, md_b1 + (size_t)e * DMODEL,
             M1, (int)BT, DMODEL, DMODEL, e > 0, 0, nullptr, EW, NEXP, e);
    prelu_ip_kernel<<<ew_blocks(BT * DMODEL), 256, 0, stream>>>(M1, (int)(BT * DMODEL), md_a1);

    for (int e = 0; e < NEXP; ++e)
        gemm(M1, md_w2 + (size_t)e * DMODEL * 129, md_b2 + (size_t)e * 129,
             M2, (int)BT, 129, DMODEL, e > 0, 0, nullptr, EW, NEXP, e);

    finalize_kernel<<<ew_blocks(BT * 161), 256, 0, stream>>>(M2, PH, out);
}
